// int8Linear_51591147160086
// MI455X (gfx1250) — matrix operand order repaired and checked by reading, not compiled, not run on hardware
//
#include <hip/hip_runtime.h>

// ---------------------------------------------------------------------------
// int8 block-quantized linear:  out[M,N] = x[M,K] * dequant(W)[N,K]^T + bias
//   M = 4*2048 = 8192, N = 4096, K = 4096, scale block = 64 along K
// Compute-bound GEMM (AI ~1.6 KFLOP/B) -> f16 WMMA path:
//   dequant W->f16 and convert x->f16 during LDS staging, accumulate f32 in
//   V_WMMA_F32_16X16X32_F16. BK=64 matches the quant block exactly.
// Software-pipelined: global loads for chunk k+1 issue before the 16-WMMA
// compute phase of chunk k, so s_wait_loadcnt lands a full phase later.
// ---------------------------------------------------------------------------

typedef _Float16 v16h __attribute__((ext_vector_type(16)));
typedef _Float16 v8h  __attribute__((ext_vector_type(8)));
typedef _Float16 v4h  __attribute__((ext_vector_type(4)));
typedef float    v8f  __attribute__((ext_vector_type(8)));
typedef float    v4f  __attribute__((ext_vector_type(4)));
typedef int      v4i  __attribute__((ext_vector_type(4)));

#define M_TOT 8192
#define N_TOT 4096
#define K_TOT 4096
#define BM 128
#define BN 128
#define BK 64
#define LDA 72   // halves per LDS row (64 data + 8 pad) -> 144B stride, bank-friendly
#define LDB 72

__global__ __launch_bounds__(256)
void int8_linear_wmma_f16(const float* __restrict__ x,
                          const int*   __restrict__ qw,
                          const float* __restrict__ scales,
                          const float* __restrict__ bias,
                          float*       __restrict__ out) {
    __shared__ _Float16 lsA[BM * LDA];   // 18432 B
    __shared__ _Float16 lsB[BN * LDB];   // 18432 B

    const int tid   = threadIdx.x;
    const int lane  = tid & 31;
    const int wave  = tid >> 5;
    const int waveM = wave & 3;          // 4 waves along M  -> 32 rows each
    const int waveN = wave >> 2;         // 2 waves along N  -> 64 cols each

    const int mBase = blockIdx.y * BM;
    const int nBase = blockIdx.x * BN;

    const int halfSel = lane >> 4;       // 0: lanes 0-15, 1: lanes 16-31
    const int l15     = lane & 15;
    const int kOffA   = halfSel * 16;    // A frag: K 0..15 vs 16..31
    const int kOffB   = halfSel * 16;    // B frag: K 0..15 vs 16..31

    v8f acc[2][4] = {};                  // 2 M-tiles x 4 N-tiles of 16x16 f32

    // staging registers for one BK=64 chunk (128x64 of A and B each)
    v4f  ax[8];
    v4i  bw[8];
    float sc[8];

    auto loadChunk = [&](int kc) {
#pragma unroll
        for (int i = 0; i < 8; ++i) {
            const int f   = i * 256 + tid;       // 2048 vec4 per tile
            const int row = f >> 4;              // 0..127
            const int c4  = f & 15;              // 0..15 (x4 elements)
            ax[i] = *(const v4f*)(x  + (size_t)(mBase + row) * K_TOT + kc + c4 * 4);
            bw[i] = *(const v4i*)(qw + (size_t)(nBase + row) * K_TOT + kc + c4 * 4);
            // one scale per (row, 64-wide K chunk): exact quant-block match
            sc[i] = scales[(size_t)(nBase + row) * (K_TOT / 64) + (kc >> 6)];
        }
    };

    auto storeChunk = [&]() {
#pragma unroll
        for (int i = 0; i < 8; ++i) {
            const int f   = i * 256 + tid;
            const int row = f >> 4;
            const int c4  = f & 15;
            v4h ha = { (_Float16)ax[i][0], (_Float16)ax[i][1],
                       (_Float16)ax[i][2], (_Float16)ax[i][3] };
            *(v4h*)(lsA + row * LDA + c4 * 4) = ha;
            v4h hb = { (_Float16)((float)bw[i][0] * sc[i]),
                       (_Float16)((float)bw[i][1] * sc[i]),
                       (_Float16)((float)bw[i][2] * sc[i]),
                       (_Float16)((float)bw[i][3] * sc[i]) };
            *(v4h*)(lsB + row * LDB + c4 * 4) = hb;
        }
    };

    loadChunk(0);                        // pipeline prologue

    for (int kc = 0; kc < K_TOT; kc += BK) {
        __syncthreads();                 // prior iteration's LDS reads done
        storeChunk();                    // convert/dequant -> LDS (f16)
        __syncthreads();

        if (kc + BK < K_TOT) {
            loadChunk(kc + BK);          // issue next chunk; waited next store
            if (kc + 2 * BK < K_TOT) {   // warm L2 for chunk k+2
                const int row = tid >> 4, c4 = tid & 15;
                __builtin_prefetch(x  + (size_t)(mBase + row) * K_TOT + kc + 2 * BK + c4 * 4, 0, 1);
                __builtin_prefetch(qw + (size_t)(nBase + row) * K_TOT + kc + 2 * BK + c4 * 4, 0, 1);
            }
        }

        // ---- 2 K-substeps x 8 WMMA (A and B both in contiguous k = 16*half + s order) --
#pragma unroll
        for (int ks = 0; ks < 2; ++ks) {
            const int ko = ks * 32;
            v16h aF[2];
#pragma unroll
            for (int mt = 0; mt < 2; ++mt) {
                const _Float16* p = lsA + (waveM * 32 + mt * 16 + l15) * LDA + ko + kOffA;
                v8h lo = *(const v8h*)(p);           // K k0..k0+7
                v8h hi = *(const v8h*)(p + 8);       // K k0+8..k0+15
                aF[mt] = __builtin_shufflevector(lo, hi,
                          0,1,2,3,4,5,6,7,8,9,10,11,12,13,14,15);
            }
#pragma unroll
            for (int nt = 0; nt < 4; ++nt) {
                const _Float16* p = lsB + (waveN * 64 + nt * 16 + l15) * LDB + ko + kOffB;
                v8h lo = *(const v8h*)(p);           // K kOffB..+7
                v8h hi = *(const v8h*)(p + 8);       // K kOffB+8..+15
                v16h bF = __builtin_shufflevector(lo, hi,
                          0,1,2,3,4,5,6,7,8,9,10,11,12,13,14,15);
#pragma unroll
                for (int mt = 0; mt < 2; ++mt) {
                    acc[mt][nt] = __builtin_amdgcn_wmma_f32_16x16x32_f16(
                        /*neg_a=*/false, aF[mt],
                        /*neg_b=*/false, bF,
                        /*c_mod=*/(short)0, acc[mt][nt],
                        /*reuse_a=*/false, /*reuse_b=*/false);
                }
            }
        }
    }

    // ---- epilogue: bias add + f32 store (C/D layout: VGPR r -> M=r|r+8) ----
#pragma unroll
    for (int nt = 0; nt < 4; ++nt) {
        const int gN = nBase + waveN * 64 + nt * 16 + l15;
        const float bv = bias[gN];
#pragma unroll
        for (int mt = 0; mt < 2; ++mt) {
#pragma unroll
            for (int r = 0; r < 8; ++r) {
                const int gM = mBase + waveM * 32 + mt * 16 + r + halfSel * 8;
                out[(size_t)gM * N_TOT + gN] = acc[mt][nt][r] + bv;
            }
        }
    }
}

extern "C" void kernel_launch(void* const* d_in, const int* in_sizes, int n_in,
                              void* d_out, int out_size, void* d_ws, size_t ws_size,
                              hipStream_t stream) {
    const float* x      = (const float*)d_in[0];   // [4,2048,4096] f32
    const int*   qw     = (const int*)  d_in[1];   // [4096,4096] int32 (int8 values)
    const float* scales = (const float*)d_in[2];   // [262144,1] f32
    const float* bias   = (const float*)d_in[3];   // [4096] f32
    float*       out    = (float*)d_out;           // [4,2048,4096] f32

    dim3 grid(N_TOT / BN, M_TOT / BM, 1);          // (32, 64)
    dim3 block(256, 1, 1);                         // 8 waves (wave32)
    int8_linear_wmma_f16<<<grid, block, 0, stream>>>(x, qw, scales, bias, out);
}
